// BahdanauAttention_16080357556505
// MI455X (gfx1250) — matrix operand order repaired and checked by reading, not compiled, not run on hardware
//
#include <hip/hip_runtime.h>
#include <hip/hip_bf16.h>

#define B_   32
#define S_   4096
#define DH_  512
#define EH_  512
#define A_   256

typedef __attribute__((ext_vector_type(16))) __bf16 v16bf;
typedef __attribute__((ext_vector_type(8)))  float  v8f;

// ---------------------------------------------------------------------------
// Kernel 1: dec_proj[b,a] = sum_k decoder_hidden[b,k] * Wd[k,a]   (32x256)
// ---------------------------------------------------------------------------
__global__ __launch_bounds__(256)
void dec_proj_kernel(const float* __restrict__ dh,
                     const float* __restrict__ Wd,
                     float* __restrict__ dp) {
    const int b = blockIdx.x;
    const int a = threadIdx.x;
    const float* h = dh + b * DH_;
    float acc = 0.f;
    #pragma unroll 8
    for (int k = 0; k < DH_; ++k)
        acc = fmaf(h[k], Wd[k * A_ + a], acc);
    dp[b * A_ + a] = acc;
}

// ---------------------------------------------------------------------------
// Kernel 2 (WMMA): for each (b, s) compute
//   energy[b,s] = mask ? v . tanh(dec_proj[b,:] + enc[b,s,:] @ We) : -inf
// Block: 512 threads = 16 waves; wave handles 16 S-rows x A=256.
// K loop: EH=512 in 16 steps of 32 (v_wmma_f32_16x16x32_bf16).
// ---------------------------------------------------------------------------
__global__ __launch_bounds__(512)
void enc_energy_kernel(const float* __restrict__ enc,
                       const float* __restrict__ We,
                       const float* __restrict__ dec_proj,
                       const float* __restrict__ vvec,
                       const int*   __restrict__ mask,
                       float* __restrict__ energy) {
    // Staged B fragments for one K-step: 16 n_tiles * 32 lanes * 16 bf16 = 16 KB
    __shared__ __attribute__((aligned(32))) __bf16 Bs[16 * 512];

    const int t    = threadIdx.x;
    const int lane = t & 31;
    const int wave = t >> 5;
    const int b     = blockIdx.x >> 4;   // 16 s-tiles of 256 rows per batch
    const int stile = blockIdx.x & 15;
    const int s0    = stile * 256 + wave * 16;

    // --- staging role: thread t fills lane-slot (t&31) of fragment (t>>5) ---
    // B (32x16, K x N) layout used here (contiguous k, not ISA table order): lanes 0-15 hold N=lane, K=0..15 in
    // elements 0..15; lanes 16-31 hold N=lane-16, K=16..31.
    const int n_st    = (t >> 5) * 16 + (t & 15);
    const int kofs_st = ((t & 31) < 16) ? 0 : 16;

    // --- A fragment addressing (contiguous k order, same as B; not ISA table order) ---
    // lanes 0-15: M=lane, K = 0..15; lanes 16-31: M=lane-16, K = 16..31.
    const int row = s0 + (lane & 15);
    const float* arow_base = enc + ((size_t)b * S_ + row) * EH_;
    const int kA = (lane < 16) ? 0 : 16;

    v8f c[16];
    #pragma unroll
    for (int n = 0; n < 16; ++n) c[n] = (v8f)(0.0f);

    for (int kt = 0; kt < 16; ++kt) {
        // Stage We slab (fp32 -> bf16, fragment-swizzled) for this K-step.
        union { v16bf v; __bf16 e[16]; } bst;
        const float* wcol = We + (size_t)(kt * 32 + kofs_st) * A_ + n_st;
        #pragma unroll
        for (int e = 0; e < 16; ++e)
            bst.e[e] = (__bf16)wcol[(size_t)e * A_];
        __syncthreads();                       // prior iteration done reading Bs
        *(v16bf*)(&Bs[t * 16]) = bst.v;        // ds_store_b128 x2
        __syncthreads();

        // Load + convert A fragment (coalesced 128b global loads).
        union { v16bf v; __bf16 e[16]; } afr;
        const float* ar = arow_base + kt * 32 + kA;
        float4 f0 = *(const float4*)(ar);
        float4 f1 = *(const float4*)(ar + 4);
        float4 f2 = *(const float4*)(ar + 8);
        float4 f3 = *(const float4*)(ar + 12);
        afr.e[0]  = (__bf16)f0.x; afr.e[1]  = (__bf16)f0.y;
        afr.e[2]  = (__bf16)f0.z; afr.e[3]  = (__bf16)f0.w;
        afr.e[4]  = (__bf16)f1.x; afr.e[5]  = (__bf16)f1.y;
        afr.e[6]  = (__bf16)f1.z; afr.e[7]  = (__bf16)f1.w;
        afr.e[8]  = (__bf16)f2.x; afr.e[9]  = (__bf16)f2.y;
        afr.e[10] = (__bf16)f2.z; afr.e[11] = (__bf16)f2.w;
        afr.e[12] = (__bf16)f3.x; afr.e[13] = (__bf16)f3.y;
        afr.e[14] = (__bf16)f3.z; afr.e[15] = (__bf16)f3.w;

        #pragma unroll
        for (int n = 0; n < 16; ++n) {
            v16bf bf = *(const v16bf*)(&Bs[(n * 32 + lane) * 16]);
            c[n] = __builtin_amdgcn_wmma_f32_16x16x32_bf16(
                false, afr.v, false, bf, (short)0, c[n], false, false);
        }
    }

    // Epilogue: esum[r] = sum over this lane's columns of tanh(c + dec)*v.
    // C/D layout: lanes 0-15: N=lane, VGPR r -> M=r; lanes 16-31: N=lane-16,
    // VGPR r -> M=r+8.
    float esum[8] = {0.f, 0.f, 0.f, 0.f, 0.f, 0.f, 0.f, 0.f};
    const float* dp = dec_proj + b * A_;
    const int cl = lane & 15;
    #pragma unroll
    for (int n = 0; n < 16; ++n) {
        const int col = n * 16 + cl;
        const float d  = dp[col];
        const float vv = vvec[col];
        #pragma unroll
        for (int r = 0; r < 8; ++r)
            esum[r] += tanhf(c[n][r] + d) * vv;
    }
    // Reduce across the 16 lanes of each half (xor stays inside the half).
    #pragma unroll
    for (int r = 0; r < 8; ++r) {
        float x = esum[r];
        x += __shfl_xor(x, 1, 32);
        x += __shfl_xor(x, 2, 32);
        x += __shfl_xor(x, 4, 32);
        x += __shfl_xor(x, 8, 32);
        esum[r] = x;
    }
    if (lane == 0 || lane == 16) {
        const int rbase = s0 + ((lane == 0) ? 0 : 8);
        #pragma unroll
        for (int r = 0; r < 8; ++r) {
            const int srow = rbase + r;
            const float ev = (mask[b * S_ + srow] == 0)
                                 ? -__builtin_inff() : esum[r];
            energy[b * S_ + srow] = ev;
        }
    }
}

// ---------------------------------------------------------------------------
// Kernel 3: row softmax over S with all-masked -> 0 (nan_to_num) guard.
// ---------------------------------------------------------------------------
__global__ __launch_bounds__(256)
void softmax_kernel(const float* __restrict__ energy,
                    float* __restrict__ attn) {
    __shared__ float red[256];
    const int b = blockIdx.x;
    const int t = threadIdx.x;
    const float* e = energy + b * S_;
    float* out = attn + b * S_;

    float m = -__builtin_inff();
    for (int s = t; s < S_; s += 256) m = fmaxf(m, e[s]);
    red[t] = m; __syncthreads();
    for (int o = 128; o > 0; o >>= 1) {
        if (t < o) red[t] = fmaxf(red[t], red[t + o]);
        __syncthreads();
    }
    const float maxv = red[0];
    __syncthreads();

    if (maxv == -__builtin_inff()) {          // fully-masked row -> zeros
        for (int s = t; s < S_; s += 256) out[s] = 0.f;
        return;
    }

    float sum = 0.f;
    for (int s = t; s < S_; s += 256) sum += __expf(e[s] - maxv);
    red[t] = sum; __syncthreads();
    for (int o = 128; o > 0; o >>= 1) {
        if (t < o) red[t] += red[t + o];
        __syncthreads();
    }
    const float inv = 1.0f / red[0];
    for (int s = t; s < S_; s += 256) out[s] = __expf(e[s] - maxv) * inv;
}

// ---------------------------------------------------------------------------
// Kernel 4: context[b,e] = sum_s attn[b,s] * enc[b,s,e]   (coalesced over e)
// ---------------------------------------------------------------------------
__global__ __launch_bounds__(512)
void context_kernel(const float* __restrict__ attn,
                    const float* __restrict__ enc,
                    float* __restrict__ ctx) {
    __shared__ float at[512];
    const int b = blockIdx.x;
    const int t = threadIdx.x;
    const float* arow  = attn + b * S_;
    const float* ebase = enc + (size_t)b * S_ * EH_;
    float acc = 0.f;
    for (int sc = 0; sc < S_; sc += 512) {
        __syncthreads();
        at[t] = arow[sc + t];
        __syncthreads();
        const float* ep = ebase + (size_t)sc * EH_ + t;
        #pragma unroll 4
        for (int j = 0; j < 512; ++j)
            acc = fmaf(at[j], ep[(size_t)j * EH_], acc);
    }
    ctx[b * EH_ + t] = acc;
}

// ---------------------------------------------------------------------------
extern "C" void kernel_launch(void* const* d_in, const int* in_sizes, int n_in,
                              void* d_out, int out_size, void* d_ws, size_t ws_size,
                              hipStream_t stream) {
    const float* dh   = (const float*)d_in[0];   // (32, 512)
    const float* enc  = (const float*)d_in[1];   // (32, 4096, 512)
    const int*   mask = (const int*)  d_in[2];   // (32, 4096)
    const float* Wd   = (const float*)d_in[3];   // (512, 256)
    const float* We   = (const float*)d_in[4];   // (512, 256)
    const float* v    = (const float*)d_in[5];   // (256,)

    float* out  = (float*)d_out;
    float* ctx  = out;                 // (32, 512)  first in return order
    float* attn = out + B_ * EH_;      // (32, 4096) second

    float* dp     = (float*)d_ws;      // dec_proj: 32*256 floats
    float* energy = dp + B_ * A_;      // energy:   32*4096 floats

    dec_proj_kernel  <<<B_,        256, 0, stream>>>(dh, Wd, dp);
    enc_energy_kernel<<<B_ * 16,   512, 0, stream>>>(enc, We, dp, v, mask, energy);
    softmax_kernel   <<<B_,        256, 0, stream>>>(energy, attn);
    context_kernel   <<<B_,        512, 0, stream>>>(attn, enc, ctx);
}
